// SelfAttention_12197707121171
// MI455X (gfx1250) — matrix operand order repaired and checked by reading, not compiled, not run on hardware
//
#include <hip/hip_runtime.h>
#include <hip/hip_bf16.h>
#include <stdint.h>

// ---------------------------------------------------------------------------
// CDNA5 (gfx1250) self-attention forward.
// GEMMs: v_wmma_f32_16x16x32_bf16, 128x128x32 block tile, 32x64 wave tile,
// double-buffered LDS fed by GLOBAL_LOAD_ASYNC_TO_LDS_B128 (ASYNCcnt).
// Band-mask structure exploited: masked logits blocks skipped, att^T@V
// K-range clamped to the nonzero band + uniform tail rows.
// ---------------------------------------------------------------------------

typedef __attribute__((ext_vector_type(16))) __bf16 v16bf;
typedef __attribute__((ext_vector_type(8)))  float  v8f;
typedef __attribute__((ext_vector_type(4)))  int    v4i;

#if defined(__has_builtin)
#if __has_builtin(__builtin_amdgcn_global_load_async_to_lds_b128) && \
    __has_builtin(__builtin_amdgcn_s_wait_asynccnt)
#define ASYNC_COPY 1
#endif
#endif

#ifdef ASYNC_COPY
typedef __attribute__((address_space(1))) v4i v4i_g;   // global (AS1)
typedef __attribute__((address_space(3))) v4i v4i_l;   // LDS (AS3)
__device__ __forceinline__ void async_ld_b128(const void* g, void* l) {
    // global -> LDS direct, tracked by ASYNCcnt (no VGPR staging)
    __builtin_amdgcn_global_load_async_to_lds_b128(
        (v4i_g*)(uintptr_t)g, (v4i_l*)(uintptr_t)l, 0, 0);
}
#endif

struct BF16Frag { uint4 lo, hi; };

__device__ __forceinline__ v16bf make_frag(uint4 lo, uint4 hi) {
    BF16Frag t{lo, hi};
    return __builtin_bit_cast(v16bf, t);
}

__device__ __forceinline__ uint16_t f32_to_bf16(float f) {
    uint32_t u = __builtin_bit_cast(uint32_t, f);
    uint32_t r = (u + 0x7FFFu + ((u >> 16) & 1u)) >> 16;   // round-to-nearest-even
    return (uint16_t)r;
}

// --------------------------- f32 -> bf16 convert ---------------------------
__global__ __launch_bounds__(256) void cvt_bf16_kernel(
    const float* __restrict__ in, uint16_t* __restrict__ out, int n)
{
    int i = blockIdx.x * blockDim.x + threadIdx.x;
    if (i < n) out[i] = f32_to_bf16(in[i]);
}

// --------------------- transpose + f32 -> bf16 convert ---------------------
// in: [R][C] f32 row-major  ->  out: [C][R] bf16 row-major
__global__ __launch_bounds__(256) void tcvt_bf16_kernel(
    const float* __restrict__ in, uint16_t* __restrict__ out, int R, int C)
{
    __shared__ float tile[32][33];
    const int c0 = blockIdx.x * 32, r0 = blockIdx.y * 32;
    const int tx = threadIdx.x, ty = threadIdx.y;          // block (32, 8)
    #pragma unroll
    for (int i = 0; i < 32; i += 8)
        tile[ty + i][tx] = in[(size_t)(r0 + ty + i) * C + c0 + tx];
    __syncthreads();
    #pragma unroll
    for (int i = 0; i < 32; i += 8)
        out[(size_t)(c0 + ty + i) * R + r0 + tx] = f32_to_bf16(tile[tx][ty + i]);
}

// ------------------------------- NT GEMM -----------------------------------
// C = scale * (A @ Bt^T)
//   A  : [M][Kd] bf16 row-major,  Bt : [N][Kd] bf16 row-major
// MODE 0: C bf16 [M][N]   MODE 1: C bf16 [N][M] (transposed)   MODE 2: C f32 [M][N]
// BSKIP : early-return blocks that are fully band-masked (logits GEMM)
// BANDED: A is att^T -> K-loop restricted to [0, band_end) U [uniform_lo, Kd)
// 256 threads = 8 wave32 arranged 4(M) x 2(N); wave tile 32x64; K-step 32.
#define TM 128
#define TN 128
#define TK 32
#define LSTRIDE 40   // TK + 8 pad: keeps 16B alignment, staggers LDS banks

template <int MODE, bool BSKIP, bool BANDED>
__global__ __launch_bounds__(256) void gemm_bf16_wmma_kernel(
    const uint16_t* __restrict__ A, const uint16_t* __restrict__ Bt,
    void* __restrict__ Cout, int M, int N, int Kd, float scale, int aperture)
{
    const int bm = blockIdx.y * TM;
    const int bn = blockIdx.x * TN;

    if (BSKIP) {   // block fully masked: softmax substitutes -FMAX, never reads it
        if (bn + TN - 1 - bm <= aperture) return;
    }

    __shared__ uint16_t As[2][TM * LSTRIDE];
    __shared__ uint16_t Bs[2][TN * LSTRIDE];

    const int tid   = threadIdx.x;
    const int wave  = tid >> 5;          // 0..7 (wave32)
    const int lane  = tid & 31;
    const int lrow  = lane & 15;
    const int kh    = lane >> 4;         // lane half selects K-striping
    const int waveM = wave >> 1;         // 0..3 -> 32-row strip
    const int waveN = wave & 1;          // 0..1 -> 64-col strip

    // K-tile ranges: [0, kend1) and [kstart2, Kd), both TK-aligned.
    int kend1 = Kd, kstart2 = Kd;
    if (BANDED) {
        // att^T row j in [bm, bm+TM): nonzero i satisfy i <= j-aperture-1,
        // i.e. i in [0, bm+TM-aperture-1), plus uniform rows i >= Kd-1-aperture.
        int e = bm + TM - aperture - 1;
        e = e < 0 ? 0 : (e > Kd ? Kd : e);
        kend1 = (e + TK - 1) / TK * TK;
        if (kend1 > Kd) kend1 = Kd;
        int u = Kd - 1 - aperture;
        u = (u < 0 ? 0 : u) / TK * TK;
        kstart2 = u < kend1 ? kend1 : u;
    }
    const int nt1 = kend1 / TK;
    const int nt2 = (Kd - kstart2) / TK;
    const int ntiles = nt1 + nt2;
    auto kt_of = [&](int it) { return it < nt1 ? it * TK : kstart2 + (it - nt1) * TK; };

    v8f acc[2][4] = {};

    // cooperative loaders: 128 rows x 32 k per matrix; thread -> (row, 32B half)
    const int row  = tid >> 1;           // 0..127
    const int half = tid & 1;            // two 16-elem (32B) chunks per row

    const uint16_t* Ag = A  + (size_t)(bm + row) * Kd + half * 16;
    const uint16_t* Bg = Bt + (size_t)(bn + row) * Kd + half * 16;
    uint16_t* Al[2] = { &As[0][row * LSTRIDE + half * 16],
                        &As[1][row * LSTRIDE + half * 16] };
    uint16_t* Bl[2] = { &Bs[0][row * LSTRIDE + half * 16],
                        &Bs[1][row * LSTRIDE + half * 16] };

    // ---- prologue: fill buffer 0 with first k-tile ----
    {
        const int k0 = kt_of(0);
#ifdef ASYNC_COPY
        async_ld_b128(Ag + k0,     Al[0]);
        async_ld_b128(Ag + k0 + 8, Al[0] + 8);
        async_ld_b128(Bg + k0,     Bl[0]);
        async_ld_b128(Bg + k0 + 8, Bl[0] + 8);
        __builtin_amdgcn_s_wait_asynccnt(0);
#else
        const uint4* ag = (const uint4*)(Ag + k0);
        const uint4* bg = (const uint4*)(Bg + k0);
        uint4 a0 = ag[0], a1 = ag[1], b0 = bg[0], b1 = bg[1];
        *(uint4*)(Al[0])     = a0;
        *(uint4*)(Al[0] + 8) = a1;
        *(uint4*)(Bl[0])     = b0;
        *(uint4*)(Bl[0] + 8) = b1;
#endif
    }
    __syncthreads();

    int buf = 0;
    for (int it = 0; it < ntiles; ++it) {
        const int  nbuf = buf ^ 1;
        const bool more = (it + 1) < ntiles;
        const int  ktn  = kt_of(more ? it + 1 : it);

        // ---- issue next k-tile into the other buffer ----
#ifdef ASYNC_COPY
        if (more) {
            async_ld_b128(Ag + ktn,     Al[nbuf]);
            async_ld_b128(Ag + ktn + 8, Al[nbuf] + 8);
            async_ld_b128(Bg + ktn,     Bl[nbuf]);
            async_ld_b128(Bg + ktn + 8, Bl[nbuf] + 8);
        }
#else
        const uint4* ag = (const uint4*)(Ag + ktn);
        const uint4* bg = (const uint4*)(Bg + ktn);
        uint4 a0 = ag[0], a1 = ag[1], b0 = bg[0], b1 = bg[1];
#endif

        // ---- compute on current buffer ----
        // A frag (16x32 bf16): lanes 0-15 hold K {0..7,16..23}, lanes 16-31
        // hold K {8..15,24..31}; row M = lane&15.
        const uint16_t* ap0 = &As[buf][(waveM * 32 + lrow) * LSTRIDE];
        const uint16_t* ap1 = ap0 + 16 * LSTRIDE;
        v16bf af0 = make_frag(*(const uint4*)(ap0 + kh * 8),
                              *(const uint4*)(ap0 + 16 + kh * 8));
        v16bf af1 = make_frag(*(const uint4*)(ap1 + kh * 8),
                              *(const uint4*)(ap1 + 16 + kh * 8));
        // B frags (32x16 bf16): column N = lane&15; lanes 0-15 K {0..7,16..23},
        // lanes 16-31 K {8..15,24..31} (same K order as A).
        v16bf bf[4];
        #pragma unroll
        for (int s = 0; s < 4; ++s) {
            const uint16_t* bp =
                &Bs[buf][(waveN * 64 + s * 16 + lrow) * LSTRIDE + kh * 8];
            bf[s] = make_frag(*(const uint4*)(bp), *(const uint4*)(bp + 16));
        }
        #pragma unroll
        for (int s = 0; s < 4; ++s) {
            acc[0][s] = __builtin_amdgcn_wmma_f32_16x16x32_bf16(
                false, af0, false, bf[s], (short)0, acc[0][s], false, false);
            acc[1][s] = __builtin_amdgcn_wmma_f32_16x16x32_bf16(
                false, af1, false, bf[s], (short)0, acc[1][s], false, false);
        }

        // ---- make next buffer visible, one barrier per k-step ----
#ifdef ASYNC_COPY
        if (more) __builtin_amdgcn_s_wait_asynccnt(0);
#else
        if (more) {
            *(uint4*)(Al[nbuf])     = a0;
            *(uint4*)(Al[nbuf] + 8) = a1;
            *(uint4*)(Bl[nbuf])     = b0;
            *(uint4*)(Bl[nbuf] + 8) = b1;
        }
#endif
        __syncthreads();
        buf = nbuf;
    }

    // D layout: VGPR v, lanes 0-15 -> (M=v, N=lane); lanes 16-31 -> (M=v+8).
    const int r0 = bm + waveM * 32 + 8 * kh;
    const int c0 = bn + waveN * 64 + lrow;
    #pragma unroll
    for (int mi = 0; mi < 2; ++mi) {
        #pragma unroll
        for (int s = 0; s < 4; ++s) {
            const int r = r0 + mi * 16;
            const int c = c0 + s * 16;
            if (MODE == 2) {
                float* p = (float*)Cout + (size_t)r * N + c;
                #pragma unroll
                for (int v = 0; v < 8; ++v)
                    p[(size_t)v * N] = acc[mi][s][v] * scale;
            } else if (MODE == 0) {
                uint16_t* p = (uint16_t*)Cout + (size_t)r * N + c;
                #pragma unroll
                for (int v = 0; v < 8; ++v)
                    p[(size_t)v * N] = f32_to_bf16(acc[mi][s][v] * scale);
            } else {
                // transposed: 8 consecutive bf16 per lane -> one 16B store
                uint16_t* p = (uint16_t*)Cout + (size_t)c * M + r;
                uint32_t w[4];
                #pragma unroll
                for (int v = 0; v < 4; ++v) {
                    uint32_t lo = f32_to_bf16(acc[mi][s][2 * v]     * scale);
                    uint32_t hi = f32_to_bf16(acc[mi][s][2 * v + 1] * scale);
                    w[v] = lo | (hi << 16);
                }
                *(uint4*)p = make_uint4(w[0], w[1], w[2], w[3]);
            }
        }
    }
}

// ------------------------------ masked softmax -----------------------------
// One block (256 thr) per row. keep iff (j - i) > aperture, else -FMAX.
// Fully-masked 256-wide chunks skip the global load entirely (uniform branch).
// Writes att (f32, to d_out) and att^T (bf16, GEMM A-operand for att^T @ V).
__global__ __launch_bounds__(256) void softmax_kernel(
    const float* __restrict__ logits, float* __restrict__ att,
    uint16_t* __restrict__ attT, int n, int aperture)
{
    __shared__ float red[256];
    const int i   = blockIdx.x;
    const int tid = threadIdx.x;
    const float FMAX = 3.4028235e38f;
    const float* row = logits + (size_t)i * n;

    float vals[16];
    float mx = -FMAX;
    #pragma unroll
    for (int t = 0; t < 16; ++t) {
        int j = tid + t * 256;
        float v = -FMAX;
        if (t * 256 + 255 - i > aperture)            // chunk has a kept column
            v = ((j - i) > aperture) ? row[j] : -FMAX;
        vals[t] = v;
        mx = fmaxf(mx, v);
    }
    red[tid] = mx; __syncthreads();
    for (int s = 128; s > 0; s >>= 1) {
        if (tid < s) red[tid] = fmaxf(red[tid], red[tid + s]);
        __syncthreads();
    }
    mx = red[0]; __syncthreads();

    float sum = 0.f;
    #pragma unroll
    for (int t = 0; t < 16; ++t) { vals[t] = __expf(vals[t] - mx); sum += vals[t]; }
    red[tid] = sum; __syncthreads();
    for (int s = 128; s > 0; s >>= 1) {
        if (tid < s) red[tid] += red[tid + s];
        __syncthreads();
    }
    float inv = 1.0f / red[0];

    #pragma unroll
    for (int t = 0; t < 16; ++t) {
        int j = tid + t * 256;
        float w = vals[t] * inv;
        att[(size_t)i * n + j]  = w;
        attT[(size_t)j * n + i] = f32_to_bf16(w);
    }
}

// ------------------------------- launcher ----------------------------------
extern "C" void kernel_launch(void* const* d_in, const int* in_sizes, int n_in,
                              void* d_out, int out_size, void* d_ws, size_t ws_size,
                              hipStream_t stream)
{
    const int n = 4096, m = 1024, d = 1024;
    const int APER = 256;
    const float* x    = (const float*)d_in[0];
    const float* WK   = (const float*)d_in[1];
    const float* WQ   = (const float*)d_in[2];
    const float* WV   = (const float*)d_in[3];
    const float* Wout = (const float*)d_in[4];

    char*  ws  = (char*)d_ws;
    size_t off = 0;
    auto alloc = [&](size_t bytes) -> char* {
        char* p = ws + off;
        off = (off + bytes + 255) & ~size_t(255);
        return p;
    };

    uint16_t* xb    = (uint16_t*)alloc((size_t)n * m * 2);   // x bf16          [n][m]
    uint16_t* Wkt   = (uint16_t*)alloc((size_t)m * d * 2);   // WK^T bf16       [d][m]
    uint16_t* Wqt   = (uint16_t*)alloc((size_t)m * d * 2);   // WQ^T bf16       [d][m]
    uint16_t* Wvt   = (uint16_t*)alloc((size_t)m * d * 2);   // WV^T bf16       [d][m]
    uint16_t* Wot   = (uint16_t*)alloc((size_t)d * m * 2);   // Wout^T bf16     [m][d]
    uint16_t* Kb    = (uint16_t*)alloc((size_t)n * d * 2);   // K bf16          [n][d]
    uint16_t* Qb    = (uint16_t*)alloc((size_t)n * d * 2);   // Q*0.06 bf16     [n][d]
    uint16_t* Vt    = (uint16_t*)alloc((size_t)d * n * 2);   // V^T bf16        [d][n]
    uint16_t* y1b   = (uint16_t*)alloc((size_t)n * d * 2);   // att^T @ V bf16  [n][d]
    float*    logit = (float*)   alloc((size_t)n * n * 4);   // logits f32      [n][n]
    uint16_t* attT  = (uint16_t*)alloc((size_t)n * n * 2);   // att^T bf16      [n][n]

    float* y_out   = (float*)d_out;                 // [n][m]
    float* att_out = (float*)d_out + (size_t)n * m; // [n][n]

    // 1) convert x to bf16
    cvt_bf16_kernel<<<(n * m + 255) / 256, 256, 0, stream>>>(x, xb, n * m);

    // 2) transpose-convert the four weight matrices
    dim3 tblk(32, 8);
    tcvt_bf16_kernel<<<dim3(d / 32, m / 32), tblk, 0, stream>>>(WK, Wkt, m, d);
    tcvt_bf16_kernel<<<dim3(d / 32, m / 32), tblk, 0, stream>>>(WQ, Wqt, m, d);
    tcvt_bf16_kernel<<<dim3(d / 32, m / 32), tblk, 0, stream>>>(WV, Wvt, m, d);
    tcvt_bf16_kernel<<<dim3(m / 32, d / 32), tblk, 0, stream>>>(Wout, Wot, d, m);

    // 3) projections: K, Q (scaled), V (written transposed)
    dim3 gProj(d / TN, n / TM);
    gemm_bf16_wmma_kernel<0, false, false><<<gProj, 256, 0, stream>>>(
        xb, Wkt, Kb, n, d, m, 1.0f, APER);
    gemm_bf16_wmma_kernel<0, false, false><<<gProj, 256, 0, stream>>>(
        xb, Wqt, Qb, n, d, m, 0.06f, APER);
    gemm_bf16_wmma_kernel<1, false, false><<<gProj, 256, 0, stream>>>(
        xb, Wvt, Vt, n, d, m, 1.0f, APER);

    // 4) logits = Q @ K^T; fully-masked 128x128 blocks early-return (~55%)
    dim3 gLog(n / TN, n / TM);
    gemm_bf16_wmma_kernel<2, true, false><<<gLog, 256, 0, stream>>>(
        Qb, Kb, logit, n, n, d, 1.0f, APER);

    // 5) banded mask + softmax; emit att (f32) and att^T (bf16)
    softmax_kernel<<<n, 256, 0, stream>>>(logit, att_out, attT, n, APER);

    // 6) y1 = att^T @ V; K-loop clamped to nonzero band + uniform tail rows
    dim3 gY1(d / TN, n / TM);
    gemm_bf16_wmma_kernel<0, false, true><<<gY1, 256, 0, stream>>>(
        attT, Vt, y1b, n, d, n, 1.0f, APER);

    // 7) y = y1 @ Wout -> f32 output
    dim3 gY2(m / TN, n / TM);
    gemm_bf16_wmma_kernel<2, false, false><<<gY2, 256, 0, stream>>>(
        y1b, Wot, y_out, n, m, d, 1.0f, APER);
}
